// PillarAttentionClassifier_76124000354424
// MI455X (gfx1250) — matrix operand order repaired and checked by reading, not compiled, not run on hardware
//
#include <hip/hip_runtime.h>
#include <hip/hip_bf16.h>

typedef __attribute__((ext_vector_type(16))) _Float16 v16h;
typedef __attribute__((ext_vector_type(8)))  float    v8f;

#define NW 4                       // waves per block (wave32)
#define ROWS_PER_BLOCK (NW * 16)   // 16 rows (WMMA M) per wave
#define MAX_BLOCKS 2048            // persistent blocks; tiles distributed grid-stride

__constant__ int cIDX[12][3] = {
  {0,1,2},{3,4,0},{5,6,0},{7,8,9},{10,11,0},{12,13,14},
  {15,16,0},{17,18,0},{19,0,0},{20,21,22},{23,24,25},{26,27,28}};
__constant__ float cMSK[12][3] = {
  {1,1,1},{1,1,0},{1,1,0},{1,1,1},{1,1,0},{1,1,1},
  {1,1,0},{1,1,0},{1,0,0},{1,1,1},{1,1,1},{1,1,1}};

struct Params {
  const float *x, *We, *be, *eg, *ebt, *em, *ev, *Wa, *ba;
  const float *W1, *b1, *g1, *bb1, *m1, *v1;
  const float *W2, *b2, *g2, *bb2, *m2, *v2;
  const float *W3, *b3, *g3, *bb3, *m3, *v3;
  const float *W4, *b4;
  float *out;
  int B;
};

__device__ __forceinline__ v8f wmma16(v16h a, v16h b, v8f c) {
  // D = A(16x32 f16) * B(32x16 f16) + C(16x16 f32)
  return __builtin_amdgcn_wmma_f32_16x16x32_f16(false, a, false, b,
                                                (short)0, c, false, false);
}

// A-fragment (16x32 f16) from LDS row-major [16][Kstride] activations.
// ISA layout: lane L -> M = L%16, g = L/16; halves 0..7 -> K = kb + g*8 + h,
// halves 8..15 -> K = kb + 16 + g*8 + (h-8). Two contiguous 16B runs.
__device__ __forceinline__ v16h load_afrag(const _Float16* A, int Kstride,
                                           int kb, int lane) {
  const int M = lane & 15, g = lane >> 4;
  const _Float16* p0 = A + M * Kstride + kb + g * 8;
  union { uint4 q[2]; v16h v; } u;
  u.q[0] = *(const uint4*)p0;
  u.q[1] = *(const uint4*)(p0 + 16);
  return u.v;
}

// B-fragment (32x16 f16) from LDS weights stored TRANSPOSED [N][Kstride].
// ISA layout: lane L -> N = L%16 (+16*ntile), g = L/16; halves 0..7 -> K = kb + g*8 + h,
// halves 8..15 -> K = kb + 16 + g*8 + (h-8). Two contiguous 16B runs.
__device__ __forceinline__ v16h load_bfrag(const _Float16* Wt, int Kstride,
                                           int kb, int ntile, int lane) {
  const int N = (lane & 15) + ntile * 16, g = lane >> 4;
  const _Float16* p = Wt + N * Kstride + kb + g * 8;
  union { uint4 q[2]; v16h v; } u;
  u.q[0] = *(const uint4*)p;
  u.q[1] = *(const uint4*)(p + 16);
  return u.v;
}

// A-fragment of attended = reps * softmax-weight (built on the fly, f32->f16)
__device__ __forceinline__ v16h make_attended(const _Float16* reps,
                                              const float* wgt, int kb, int lane) {
  const int M = lane & 15, g = lane >> 4;
  const _Float16* r = reps + M * 96;
  const float* w = wgt + M * 12;
  v16h a;
#pragma unroll
  for (int h = 0; h < 16; ++h) {
    const int k = kb + g * 8 + (h < 8 ? h : 8 + (h - 8) + 16);
    a[h] = (_Float16)((float)r[k] * w[k >> 3]);   // k>>3 = encoder group index
  }
  return a;
}

__global__ __launch_bounds__(NW * 32)
void pillar_attention_fused(Params p) {
  // ---- weights in LDS, transposed [N][K], f16 (resident for whole block) ----
  __shared__ __align__(16) _Float16 sWa[16][96];   // 96x12 padded to N=16
  __shared__ __align__(16) _Float16 sW1[64][96];
  __shared__ __align__(16) _Float16 sW2[32][64];
  __shared__ __align__(16) _Float16 sW3[16][32];
  // ---- folded BN (scale/shift) + biases, f32 ----
  __shared__ float sWe[288], sBe[96], sEncS[96], sEncT[96], sBa[16];
  __shared__ float sB1[64], sS1[64], sT1[64];
  __shared__ float sB2[32], sS2[32], sT2[32];
  __shared__ float sB3[16], sS3[16], sT3[16];
  __shared__ float sW4[16], sB4[1];
  // ---- per-wave activation staging (wave-private; same-wave DS is in-order) --
  __shared__ __align__(16) _Float16 sReps[NW][16][96];
  __shared__ float                  sLog [NW][16][16];
  __shared__ float                  sWgt [NW][16][12];
  __shared__ __align__(16) _Float16 sH1  [NW][16][64];
  __shared__ __align__(16) _Float16 sH2  [NW][16][32];
  __shared__ float                  sH3  [NW][16][16];

  const int tid  = threadIdx.x;
  const int lane = tid & 31;
  const int wave = tid >> 5;
  const int g    = lane >> 4;   // half-wave group

  // ================= stage weights / params (ONCE per persistent block) =====
  for (int t = tid; t < 16 * 96; t += NW * 32) {
    int n = t / 96, k = t % 96;
    sWa[n][k] = (_Float16)(n < 12 ? p.Wa[k * 12 + n] : 0.f);
  }
  for (int t = tid; t < 64 * 96; t += NW * 32) {
    int n = t / 96, k = t % 96; sW1[n][k] = (_Float16)p.W1[k * 64 + n];
  }
  for (int t = tid; t < 32 * 64; t += NW * 32) {
    int n = t / 64, k = t % 64; sW2[n][k] = (_Float16)p.W2[k * 32 + n];
  }
  for (int t = tid; t < 16 * 32; t += NW * 32) {
    int n = t / 32, k = t % 32; sW3[n][k] = (_Float16)p.W3[k * 16 + n];
  }
  for (int t = tid; t < 288; t += NW * 32) sWe[t] = p.We[t];
  for (int t = tid; t < 96; t += NW * 32) {
    float s = p.eg[t] * rsqrtf(p.ev[t] + 1e-5f);
    sEncS[t] = s; sEncT[t] = p.ebt[t] - p.em[t] * s; sBe[t] = p.be[t];
  }
  for (int t = tid; t < 16; t += NW * 32) sBa[t] = (t < 12) ? p.ba[t] : 0.f;
  for (int t = tid; t < 64; t += NW * 32) {
    float s = p.g1[t] * rsqrtf(p.v1[t] + 1e-5f);
    sS1[t] = s; sT1[t] = p.bb1[t] - p.m1[t] * s; sB1[t] = p.b1[t];
  }
  for (int t = tid; t < 32; t += NW * 32) {
    float s = p.g2[t] * rsqrtf(p.v2[t] + 1e-5f);
    sS2[t] = s; sT2[t] = p.bb2[t] - p.m2[t] * s; sB2[t] = p.b2[t];
  }
  for (int t = tid; t < 16; t += NW * 32) {
    float s = p.g3[t] * rsqrtf(p.v3[t] + 1e-5f);
    sS3[t] = s; sT3[t] = p.bb3[t] - p.m3[t] * s; sB3[t] = p.b3[t];
    sW4[t] = p.W4[t];
  }
  if (tid == 0) sB4[0] = p.b4[0];
  __syncthreads();

  // ================= persistent grid-stride loop over 64-row tiles ==========
  const int ntiles = (p.B + ROWS_PER_BLOCK - 1) / ROWS_PER_BLOCK;
  for (int tile = blockIdx.x; tile < ntiles; tile += gridDim.x) {
    const int row0 = tile * ROWS_PER_BLOCK + wave * 16;

    // ---- encoders: Linear(3->8)+ReLU+BN, VALU (K too small for WMMA) ----
    {
      const int row = lane & 15;
      const int gr  = min(row0 + row, p.B - 1);
      const float* xr = p.x + (size_t)gr * 29;
      const int gbase = g * 6;               // 2 lanes per row, 6 groups each
#pragma unroll
      for (int q = 0; q < 6; ++q) {
        const int gg = gbase + q;
        const float x0 = xr[cIDX[gg][0]] * cMSK[gg][0];
        const float x1 = xr[cIDX[gg][1]] * cMSK[gg][1];
        const float x2 = xr[cIDX[gg][2]] * cMSK[gg][2];
        const float* w = sWe + gg * 24;      // [3][8]
#pragma unroll
        for (int e = 0; e < 8; ++e) {
          const int o = gg * 8 + e;
          float v = sBe[o] + x0 * w[e] + x1 * w[8 + e] + x2 * w[16 + e];
          v = fmaxf(v, 0.f) * sEncS[o] + sEncT[o];
          sReps[wave][row][o] = (_Float16)v;
        }
      }
    }

    // ---- attention logits: [16x96] @ [96x16pad], 3 WMMA ----
    {
      v8f acc = {};
#pragma unroll
      for (int kb = 0; kb < 3; ++kb) {
        v16h a = load_afrag(&sReps[wave][0][0], 96, kb * 32, lane);
        v16h b = load_bfrag(&sWa[0][0], 96, kb * 32, 0, lane);
        acc = wmma16(a, b, acc);
      }
      const int N = lane & 15;
#pragma unroll
      for (int r = 0; r < 8; ++r)
        sLog[wave][r + 8 * g][N] = acc[r] + sBa[N];
    }
    // softmax over 12 logits, one lane per row (reconverges before next WMMA)
    if (lane < 16) {
      const int rr = lane;
      float m = -3.4e38f;
#pragma unroll
      for (int j = 0; j < 12; ++j) m = fmaxf(m, sLog[wave][rr][j]);
      float e[12], s = 0.f;
#pragma unroll
      for (int j = 0; j < 12; ++j) { e[j] = __expf(sLog[wave][rr][j] - m); s += e[j]; }
      const float inv = 1.f / s;
#pragma unroll
      for (int j = 0; j < 12; ++j) sWgt[wave][rr][j] = e[j] * inv;
    }

    // ---- MLP1: attended[16x96] @ W1[96x64], 12 WMMA ----
    {
      v16h a0 = make_attended(&sReps[wave][0][0], &sWgt[wave][0][0],  0, lane);
      v16h a1 = make_attended(&sReps[wave][0][0], &sWgt[wave][0][0], 32, lane);
      v16h a2 = make_attended(&sReps[wave][0][0], &sWgt[wave][0][0], 64, lane);
#pragma unroll
      for (int t = 0; t < 4; ++t) {
        v8f c = {};
        c = wmma16(a0, load_bfrag(&sW1[0][0], 96,  0, t, lane), c);
        c = wmma16(a1, load_bfrag(&sW1[0][0], 96, 32, t, lane), c);
        c = wmma16(a2, load_bfrag(&sW1[0][0], 96, 64, t, lane), c);
        const int n = t * 16 + (lane & 15);
#pragma unroll
        for (int r = 0; r < 8; ++r) {
          float v = c[r] + sB1[n];
          v = fmaxf(v, 0.f) * sS1[n] + sT1[n];
          sH1[wave][r + 8 * g][n] = (_Float16)v;
        }
      }
    }

    // ---- MLP2: [16x64] @ W2[64x32], 4 WMMA ----
    {
      v16h h0 = load_afrag(&sH1[wave][0][0], 64,  0, lane);
      v16h h1 = load_afrag(&sH1[wave][0][0], 64, 32, lane);
#pragma unroll
      for (int t = 0; t < 2; ++t) {
        v8f c = {};
        c = wmma16(h0, load_bfrag(&sW2[0][0], 64,  0, t, lane), c);
        c = wmma16(h1, load_bfrag(&sW2[0][0], 64, 32, t, lane), c);
        const int n = t * 16 + (lane & 15);
#pragma unroll
        for (int r = 0; r < 8; ++r) {
          float v = c[r] + sB2[n];
          v = fmaxf(v, 0.f) * sS2[n] + sT2[n];
          sH2[wave][r + 8 * g][n] = (_Float16)v;
        }
      }
    }

    // ---- MLP3: [16x32] @ W3[32x16], 1 WMMA ----
    {
      v16h h = load_afrag(&sH2[wave][0][0], 32, 0, lane);
      v8f c = {};
      c = wmma16(h, load_bfrag(&sW3[0][0], 32, 0, 0, lane), c);
      const int n = lane & 15;
#pragma unroll
      for (int r = 0; r < 8; ++r) {
        float v = c[r] + sB3[n];
        v = fmaxf(v, 0.f) * sS3[n] + sT3[n];
        sH3[wave][r + 8 * g][n] = v;
      }
    }

    // ---- head: [16] dot W4 + sigmoid ----
    if (lane < 16) {
      const int rr = lane;
      float acc = sB4[0];
#pragma unroll
      for (int j = 0; j < 16; ++j) acc += sH3[wave][rr][j] * sW4[j];
      const float o = 1.f / (1.f + __expf(-acc));
      const int grow = row0 + rr;
      if (grow < p.B) p.out[grow] = o;
    }
  }
}

extern "C" void kernel_launch(void* const* d_in, const int* in_sizes, int n_in,
                              void* d_out, int out_size, void* d_ws, size_t ws_size,
                              hipStream_t stream) {
  (void)n_in; (void)d_ws; (void)ws_size; (void)out_size;
  Params p;
  p.x   = (const float*)d_in[0];
  p.We  = (const float*)d_in[1];
  p.be  = (const float*)d_in[2];
  p.eg  = (const float*)d_in[3];
  p.ebt = (const float*)d_in[4];
  p.em  = (const float*)d_in[5];
  p.ev  = (const float*)d_in[6];
  p.Wa  = (const float*)d_in[7];
  p.ba  = (const float*)d_in[8];
  p.W1  = (const float*)d_in[9];
  p.b1  = (const float*)d_in[10];
  p.g1  = (const float*)d_in[11];
  p.bb1 = (const float*)d_in[12];
  p.m1  = (const float*)d_in[13];
  p.v1  = (const float*)d_in[14];
  p.W2  = (const float*)d_in[15];
  p.b2  = (const float*)d_in[16];
  p.g2  = (const float*)d_in[17];
  p.bb2 = (const float*)d_in[18];
  p.m2  = (const float*)d_in[19];
  p.v2  = (const float*)d_in[20];
  p.W3  = (const float*)d_in[21];
  p.b3  = (const float*)d_in[22];
  p.g3  = (const float*)d_in[23];
  p.bb3 = (const float*)d_in[24];
  p.m3  = (const float*)d_in[25];
  p.v3  = (const float*)d_in[26];
  p.W4  = (const float*)d_in[27];
  p.b4  = (const float*)d_in[28];
  p.out = (float*)d_out;
  p.B   = in_sizes[0] / 29;

  const int ntiles = (p.B + ROWS_PER_BLOCK - 1) / ROWS_PER_BLOCK;
  const int blocks = ntiles < MAX_BLOCKS ? ntiles : MAX_BLOCKS;
  hipLaunchKernelGGL(pillar_attention_fused, dim3(blocks), dim3(NW * 32), 0, stream, p);
}
